// SqueezedCrossAttentionLayer_30571577213382
// MI455X (gfx1250) — matrix operand order repaired and checked by reading, not compiled, not run on hardware
//
#include <hip/hip_runtime.h>
#include <hip/hip_bf16.h>

// ---------------------------------------------------------------------------
// SqueezedCrossAttentionLayer for MI455X (gfx1250, wave32, WMMA bf16)
//   d=256, H=8, hd=32, B=8, Lq=256, Ls=4096
//   1) proj (templated USE_PE/MODE, 16x64 tile/wave): Q,K row-major bf16,
//      V transposed per (scene,head) as Vt[hd][Ls] bf16
//   2) flash attention, 64 keys/iter, 8 WMMAs/iter; softmax reductions done
//      in-lane after an LDS transpose (only ~10 cross-lane ops per iter)
//   3) out-proj + bias + residual + LayerNorm (WMMA + batched butterflies)
// ---------------------------------------------------------------------------

typedef __attribute__((ext_vector_type(16))) __bf16          v16bf;
typedef __attribute__((ext_vector_type(8)))  float           v8f;
typedef __attribute__((ext_vector_type(16))) unsigned short  v16u;
typedef __attribute__((ext_vector_type(8)))  unsigned short  v8u;

#define D        256
#define NHEAD    8
#define HD       32
#define NB       8
#define LQ       256
#define LS       4096
#define NQ       (NB * LQ)      // 2048
#define NS       (NB * LS)      // 32768

__device__ __forceinline__ unsigned short f2bf(float f) {
    union { float f; unsigned int u; } c; c.f = f;
    unsigned int x = c.u;
    unsigned int r = x + 0x7FFFu + ((x >> 16) & 1u);   // round-to-nearest-even
    return (unsigned short)(r >> 16);
}

__device__ __forceinline__ v16bf as_bf(v16u u) {
    union { v16u u; v16bf b; } c; c.u = u; return c.b;
}

__device__ __forceinline__ v16bf ld_bf16x16(const unsigned short* p) {
    union { v16u u; v16bf b; } c;
    c.u = *(const v16u*)p;                             // 32B (2x b128)
    return c.b;
}

__device__ __forceinline__ v8f wmma_bf16(v16bf a, v16bf b, v8f c) {
    return __builtin_amdgcn_wmma_f32_16x16x32_bf16(
        /*neg_a=*/false, a, /*neg_b=*/false, b,
        /*c_mod=*/(short)0, c, /*reuse_a=*/false, /*reuse_b=*/false);
}

// Build bf16 B-fragment (lane = column, 16 contiguous K f32 -> bf16)
__device__ __forceinline__ v16bf bfrag_from_f32(const float* pb) {
    v8f c0 = *(const v8f*)pb;
    v8f c1 = *(const v8f*)(pb + 8);
    v16u u;
    #pragma unroll
    for (int j = 0; j < 8; ++j) { u[j] = f2bf(c0[j]); u[8 + j] = f2bf(c1[j]); }
    return as_bf(u);
}

// ---------------------------------------------------------------------------
// Kernel 1: Y = (X [+ PE]) @ W^T, +bias, *scale, bf16 out.
// One wave computes a 16x64 tile (4 accumulators share one A fragment).
// MODE 0: row-major y[token][o].  MODE 1: transposed Vt[(scene*8+h)*32+hd][key]
// ---------------------------------------------------------------------------
template <bool USE_PE, int MODE>
__global__ void sca_proj_kernel(const float* __restrict__ x,
                                const float* __restrict__ pe,
                                const float* __restrict__ w,
                                const float* __restrict__ bias,
                                float scale,
                                unsigned short* __restrict__ y)
{
    const int lane = threadIdx.x & 31;
    const int wave = threadIdx.x >> 5;
    const int tile = blockIdx.x * (blockDim.x >> 5) + wave;
    const int rowTile = tile >> 2;            // 4 col-groups of 64
    const int colGrp  = tile & 3;
    const int ln16 = lane & 15;
    const int hi   = lane >> 4;

    const int row = rowTile * 16 + ln16;      // A row for this lane
    const int o0c = colGrp * 64 + ln16;       // first of 4 output cols

    v8f acc[4];
    #pragma unroll
    for (int t = 0; t < 4; ++t) acc[t] = (v8f){};

    for (int kk = 0; kk < D; kk += 32) {
        // A fragment: contiguous [kk+16*hi, +16), b128 loads
        const float* pa = x + (size_t)row * D + kk + 16 * hi;
        v8f a0 = *(const v8f*)pa;
        v8f a1 = *(const v8f*)(pa + 8);
        if (USE_PE) {
            const float* pp = pe + (size_t)row * D + kk + 16 * hi;
            a0 += *(const v8f*)pp;
            a1 += *(const v8f*)(pp + 8);
        }
        v16u au;
        #pragma unroll
        for (int j = 0; j < 8; ++j) { au[j] = f2bf(a0[j]); au[8 + j] = f2bf(a1[j]); }
        v16bf a = as_bf(au);

        #pragma unroll
        for (int t = 0; t < 4; ++t) {
            const float* pb = w + (size_t)(o0c + 16 * t) * D + kk + 16 * hi;
            acc[t] = wmma_bf16(a, bfrag_from_f32(pb), acc[t]);
        }
    }

    #pragma unroll
    for (int t = 0; t < 4; ++t) {
        const int o = o0c + 16 * t;
        const float bo = bias[o];
        #pragma unroll
        for (int j = 0; j < 8; ++j) acc[t][j] = (acc[t][j] + bo) * scale;

        if (MODE == 0) {
            #pragma unroll
            for (int j = 0; j < 8; ++j) {
                int r = rowTile * 16 + j + 8 * hi;
                y[(size_t)r * D + o] = f2bf(acc[t][j]);
            }
        } else {
            const int h = o >> 5, hd = o & 31;
            const int token0 = rowTile * 16 + 8 * hi;   // rows j consecutive
            const int scene  = token0 >> 12;
            const int key0   = token0 & (LS - 1);
            v8u ov;
            #pragma unroll
            for (int j = 0; j < 8; ++j) ov[j] = f2bf(acc[t][j]);
            *reinterpret_cast<v8u*>(
                y + ((size_t)(scene * NHEAD + h) * HD + hd) * LS + key0) = ov;
        }
    }
}

// ---------------------------------------------------------------------------
// Kernel 2: flash attention. grid = B*H*(LQ/16) = 1024 single-wave blocks.
// 64 keys per iteration: 4 QK^T WMMAs + 4 P*V WMMAs.
// Softmax: transpose S through LDS, reduce in-lane; per-lane (m,l) state for
// row = lane%16; only 2 xor-shuffles + 8 broadcast shuffles per iteration.
// ---------------------------------------------------------------------------
__global__ void sca_attn_kernel(const unsigned short* __restrict__ Qbf,
                                const unsigned short* __restrict__ Kbf,
                                const unsigned short* __restrict__ Vt,
                                float* __restrict__ Oacc)
{
    __shared__ __align__(32) float ldsS[16][64];

    const int lane = threadIdx.x & 31;
    const int ln16 = lane & 15, hi = lane >> 4;
    const int tile = blockIdx.x;
    const int qt = tile & 15;
    const int h  = (tile >> 4) & (NHEAD - 1);
    const int b  = tile >> 7;

    // Q A-fragment (1/sqrt(hd) folded into Q projection)
    const int qrow = b * LQ + qt * 16 + ln16;
    v16bf aq;
    {
        const unsigned short* pq = Qbf + (size_t)qrow * D + h * HD + 16 * hi;
        v16u t;
        #pragma unroll
        for (int j = 0; j < 8; ++j) { t[j] = pq[j]; t[8 + j] = pq[8 + j]; }
        aq = as_bf(t);
    }

    // per-lane softmax state for row = ln16 (duplicated across halves)
    float mrow = -3.0e38f, lrow = 0.0f;
    v8f o0 = {}, o1 = {};

    const unsigned short* kbase = Kbf + (size_t)b * LS * D + h * HD + hi * 16;
    const unsigned short* vbase = Vt + (size_t)(b * NHEAD + h) * HD * LS;

    for (int kc = 0; kc < LS; kc += 64) {
        // prefetch next chunk's K rows / V rows into L2 (global_prefetch_b8)
        if (kc + 64 < LS) {
            __builtin_prefetch(kbase + (size_t)(kc + 64 + ln16) * D, 0, 1);
            __builtin_prefetch(kbase + (size_t)(kc + 96 + ln16) * D, 0, 1);
            __builtin_prefetch(vbase + (size_t)(ln16 + 16 * hi) * LS + kc + 64, 0, 1);
        }

        // --- scores S = Q x K^T for 64 keys (4 x 16-key fragments)
        v8f s[4];
        v8f z = {};
        #pragma unroll
        for (int f = 0; f < 4; ++f) {
            v16bf bk = ld_bf16x16(kbase + (size_t)(kc + 16 * f + ln16) * D);
            s[f] = wmma_bf16(aq, bk, z);
        }

        // --- transpose S (C-layout: lane=key, rows j+8*hi) into row-major LDS
        __syncthreads();
        #pragma unroll
        for (int f = 0; f < 4; ++f)
            #pragma unroll
            for (int j = 0; j < 8; ++j)
                ldsS[j + 8 * hi][16 * f + ln16] = s[f][j];
        __syncthreads();

        // --- each lane reads row ln16's scores in A-fragment order:
        //     chunk c keys {32c+16hi+0..15}
        float r0[16], r1[16];
        #pragma unroll
        for (int j = 0; j < 8; ++j) {
            r0[j]     = ldsS[ln16][16 * hi + j];
            r0[8 + j] = ldsS[ln16][16 * hi + 8 + j];
            r1[j]     = ldsS[ln16][32 + 16 * hi + j];
            r1[8 + j] = ldsS[ln16][40 + 16 * hi + j];
        }

        // --- in-lane row max over this half, one xor-16 to combine halves
        float mx = -3.0e38f;
        #pragma unroll
        for (int j = 0; j < 16; ++j) mx = fmaxf(mx, fmaxf(r0[j], r1[j]));
        mx = fmaxf(mx, __shfl_xor(mx, 16));

        const float mn = fmaxf(mrow, mx);
        const float cs = __expf(mrow - mn);
        mrow = mn;

        // --- exponentiate in-lane, build P A-fragments directly, in-lane sum
        v16u pu0, pu1;
        float ps = 0.0f;
        #pragma unroll
        for (int j = 0; j < 16; ++j) {
            float e0 = __expf(r0[j] - mn);
            float e1 = __expf(r1[j] - mn);
            ps += e0 + e1;
            pu0[j] = f2bf(e0);
            pu1[j] = f2bf(e1);
        }
        ps += __shfl_xor(ps, 16);
        lrow = lrow * cs + ps;
        v16bf ap0 = as_bf(pu0);
        v16bf ap1 = as_bf(pu1);

        // --- broadcast cs into C-distribution (row j+8*hi lives in lane j+8*hi)
        float csd[8];
        #pragma unroll
        for (int j = 0; j < 8; ++j) csd[j] = __shfl(cs, j + 8 * hi);
        #pragma unroll
        for (int j = 0; j < 8; ++j) { o0[j] *= csd[j]; o1[j] *= csd[j]; }

        // --- O += P x V; Vt rows are head-dims -> contiguous 16-key loads
        {
            v16bf bv00 = ld_bf16x16(vbase + (size_t)ln16 * LS + kc + 16 * hi);
            v16bf bv01 = ld_bf16x16(vbase + (size_t)(16 + ln16) * LS + kc + 16 * hi);
            v16bf bv10 = ld_bf16x16(vbase + (size_t)ln16 * LS + kc + 32 + 16 * hi);
            v16bf bv11 = ld_bf16x16(vbase + (size_t)(16 + ln16) * LS + kc + 32 + 16 * hi);
            o0 = wmma_bf16(ap0, bv00, o0);
            o1 = wmma_bf16(ap0, bv01, o1);
            o0 = wmma_bf16(ap1, bv10, o0);
            o1 = wmma_bf16(ap1, bv11, o1);
        }
    }

    // --- finalize: bring 1/l into C-distribution, store f32
    const float invl = 1.0f / lrow;
    float invd[8];
    #pragma unroll
    for (int j = 0; j < 8; ++j) invd[j] = __shfl(invl, j + 8 * hi);

    float* obase = Oacc + (size_t)(b * LQ + qt * 16) * D + h * HD;
    #pragma unroll
    for (int j = 0; j < 8; ++j) {
        int r = j + 8 * hi;
        obase[(size_t)r * D + ln16]      = o0[j] * invd[j];
        obase[(size_t)r * D + 16 + ln16] = o1[j] * invd[j];
    }
}

// ---------------------------------------------------------------------------
// Kernel 3: Y = O @ out_w^T + out_b + residual(query), then LayerNorm.
// One wave handles 16 rows x all 256 cols (16 f32 accumulator fragments).
// ---------------------------------------------------------------------------
__global__ void sca_outln_kernel(const float* __restrict__ Oacc,
                                 const float* __restrict__ out_w,
                                 const float* __restrict__ out_b,
                                 const float* __restrict__ query,
                                 const float* __restrict__ ln_w,
                                 const float* __restrict__ ln_b,
                                 float* __restrict__ out)
{
    const int lane = threadIdx.x & 31;
    const int wave = threadIdx.x >> 5;
    const int rowTile = blockIdx.x * (blockDim.x >> 5) + wave;   // 128 total
    const int ln16 = lane & 15, hi = lane >> 4;
    const int arow = rowTile * 16 + ln16;

    v8f acc[16];
    #pragma unroll
    for (int t = 0; t < 16; ++t) acc[t] = (v8f){};

    for (int kk = 0; kk < D; kk += 32) {
        const float* pa = Oacc + (size_t)arow * D + kk + 16 * hi;
        v8f a0 = *(const v8f*)pa;
        v8f a1 = *(const v8f*)(pa + 8);
        v16u au;
        #pragma unroll
        for (int j = 0; j < 8; ++j) { au[j] = f2bf(a0[j]); au[8 + j] = f2bf(a1[j]); }
        v16bf a = as_bf(au);
        #pragma unroll
        for (int t = 0; t < 16; ++t) {
            const float* pb = out_w + (size_t)(t * 16 + ln16) * D + kk + 16 * hi;
            acc[t] = wmma_bf16(a, bfrag_from_f32(pb), acc[t]);
        }
    }

    // bias + residual, accumulate row statistics
    float sum[8], sq[8];
    #pragma unroll
    for (int j = 0; j < 8; ++j) { sum[j] = 0.0f; sq[j] = 0.0f; }
    #pragma unroll
    for (int t = 0; t < 16; ++t) {
        const int o = t * 16 + ln16;
        const float ob = out_b[o];
        #pragma unroll
        for (int j = 0; j < 8; ++j) {
            int r = rowTile * 16 + j + 8 * hi;
            float v = acc[t][j] + ob + query[(size_t)r * D + o];
            acc[t][j] = v;
            sum[j] += v;
            sq[j]  += v * v;
        }
    }
    // batched butterflies: 16 shuffles per step issued together
    #pragma unroll
    for (int m = 1; m < 16; m <<= 1) {
        float ts[8], tq[8];
        #pragma unroll
        for (int j = 0; j < 8; ++j) {
            ts[j] = __shfl_xor(sum[j], m);
            tq[j] = __shfl_xor(sq[j], m);
        }
        #pragma unroll
        for (int j = 0; j < 8; ++j) { sum[j] += ts[j]; sq[j] += tq[j]; }
    }
    float mu[8], rstd[8];
    #pragma unroll
    for (int j = 0; j < 8; ++j) {
        float mean = sum[j] * (1.0f / D);
        float var  = sq[j] * (1.0f / D) - mean * mean;
        mu[j] = mean;
        rstd[j] = rsqrtf(var + 1e-5f);
    }
    #pragma unroll
    for (int t = 0; t < 16; ++t) {
        const int o = t * 16 + ln16;
        const float gw = ln_w[o], gb = ln_b[o];
        #pragma unroll
        for (int j = 0; j < 8; ++j) {
            int r = rowTile * 16 + j + 8 * hi;
            out[(size_t)r * D + o] = (acc[t][j] - mu[j]) * rstd[j] * gw + gb;
        }
    }
}

// ---------------------------------------------------------------------------
extern "C" void kernel_launch(void* const* d_in, const int* in_sizes, int n_in,
                              void* d_out, int out_size, void* d_ws, size_t ws_size,
                              hipStream_t stream) {
    const float* source = (const float*)d_in[0];
    const float* query  = (const float*)d_in[1];
    const float* src_pe = (const float*)d_in[2];
    const float* q_pe   = (const float*)d_in[3];
    const float* in_w   = (const float*)d_in[4];
    const float* in_b   = (const float*)d_in[5];
    const float* out_w  = (const float*)d_in[6];
    const float* out_b  = (const float*)d_in[7];
    const float* ln_w   = (const float*)d_in[8];
    const float* ln_b   = (const float*)d_in[9];

    // workspace carve-up
    unsigned short* Qbf = (unsigned short*)d_ws;           // NQ*D bf16  (1 MB)
    unsigned short* Kbf = Qbf + (size_t)NQ * D;            // NS*D bf16  (16 MB)
    unsigned short* Vt  = Kbf + (size_t)NS * D;            // NS*D bf16  (16 MB, transposed)
    float*          Oacc = (float*)(Vt + (size_t)NS * D);  // NQ*D f32   (2 MB)

    const float qscale = 0.17677669529663688f;  // 1/sqrt(32)

    // projections: 16x64 tile per wave, 8 waves per 256-thread block
    sca_proj_kernel<true, 0><<<(NQ / 16) * 4 / 8, 256, 0, stream>>>(
        query, q_pe, in_w, in_b, qscale, Qbf);
    sca_proj_kernel<true, 0><<<(NS / 16) * 4 / 8, 256, 0, stream>>>(
        source, src_pe, in_w + D * D, in_b + D, 1.0f, Kbf);
    sca_proj_kernel<false, 1><<<(NS / 16) * 4 / 8, 256, 0, stream>>>(
        source, nullptr, in_w + 2 * D * D, in_b + 2 * D, 1.0f, Vt);

    // flash attention: one wave per (scene, head, 16-query tile)
    sca_attn_kernel<<<NB * NHEAD * (LQ / 16), 32, 0, stream>>>(Qbf, Kbf, Vt, Oacc);

    // out-proj + residual + LayerNorm
    sca_outln_kernel<<<(NQ / 16) / 8, 256, 0, stream>>>(
        Oacc, out_w, out_b, query, ln_w, ln_b, (float*)d_out);
}
